// VanillaRNN_40132174414447
// MI455X (gfx1250) — matrix operand order repaired and checked by reading, not compiled, not run on hardware
//
#include <hip/hip_runtime.h>
#include <hip/hip_bf16.h>
#include <math.h>

#define SEQ 512
#define IN_DIM 256
#define HID 2048
#define CLS 1024
#define BATCH 256

#define CHUNK_K 128
#define APAD 8
#define AROW (CHUNK_K + APAD)   // 136 bf16 per padded LDS row
#define NCHUNK_H (HID / CHUNK_K)     // 16
#define NCHUNK_X (IN_DIM / CHUNK_K)  // 2
#define NCHUNK_T (NCHUNK_H + NCHUNK_X)

typedef __attribute__((ext_vector_type(16))) __bf16 v16bf;
typedef __attribute__((ext_vector_type(8)))  __bf16 v8bf;
typedef __attribute__((ext_vector_type(8)))  float  v8f;

union V16BF { v16bf v; __bf16 e[16]; };

// ---------------------------------------------------------------------------
// CDNA5 async global->LDS staging (ASYNCcnt path)
// ---------------------------------------------------------------------------

// Stage 32 bytes per thread: two async b128 loads. INST_OFFSET applies to both
// the global and LDS side, so one (lds,global) address pair covers both halves.
__device__ __forceinline__ void async_stage32(unsigned lds_off, const __bf16* g) {
    unsigned long long ga = (unsigned long long)(uintptr_t)g;
    asm volatile("global_load_async_to_lds_b128 %0, %1, off\n\t"
                 "global_load_async_to_lds_b128 %0, %1, off offset:16"
                 :: "v"(lds_off), "v"(ga)
                 : "memory");
}

__device__ __forceinline__ void wait_async0() {
    asm volatile("s_wait_asynccnt 0x0" ::: "memory");
}

// ---------------------------------------------------------------------------
// Fragment helpers
// ---------------------------------------------------------------------------

// A-fragment (16x32 bf16, contiguous k = 16*(lane/16) + s) read from a padded LDS tile.
__device__ __forceinline__ v16bf lds_a_frag(const __bf16* As, int row_base, int ks, int lane) {
    const int akoff = (lane & 16) ? 16 : 0;
    const __bf16* ap = As + (row_base + (lane & 15)) * AROW + ks + akoff;
    V16BF u;
    *reinterpret_cast<v8bf*>(&u.e[0]) = *reinterpret_cast<const v8bf*>(ap);
    *reinterpret_cast<v8bf*>(&u.e[8]) = *reinterpret_cast<const v8bf*>(ap + 8);
    return u.v;
}

// B-fragment (32x16 bf16): W stored transposed [N][K].
__device__ __forceinline__ v16bf glb_b_frag(const __bf16* __restrict__ rowT, int kt, int lane) {
    const int bko = (lane & 16) ? 16 : 0;
    return *reinterpret_cast<const v16bf*>(rowT + kt + bko);
}

// ---------------------------------------------------------------------------
// One recurrence step: h_out = tanh(x_t @ Whx + Bh + h_in @ Whh)
// Block = 256 threads = 8 waves. Block tile: 32M x 256N. Wave tile: 32M x 32N.
// grid = (HID/256, BATCH/32) = (8, 8)
// ---------------------------------------------------------------------------
__global__ void __launch_bounds__(256)
rnn_step_kernel(const __bf16* __restrict__ x_bf,   // [BATCH][SEQ][IN_DIM]
                const __bf16* __restrict__ WhxT,   // [HID][IN_DIM]
                const __bf16* __restrict__ WhhT,   // [HID][HID]
                const float*  __restrict__ Bh,
                const __bf16* __restrict__ h_in,   // [BATCH][HID]
                __bf16*       __restrict__ h_out,  // [BATCH][HID]
                int t) {
    __shared__ __bf16 As[2][32 * AROW];

    const int tid   = threadIdx.x;
    const int lane  = tid & 31;
    const int wave  = tid >> 5;
    const int nlane = lane & 15;
    const int m0    = blockIdx.y * 32;
    const int nw    = blockIdx.x * 256 + wave * 32;

    v8f acc[2][2] = {};

    const __bf16* bptr0 = WhhT + (size_t)(nw + nlane) * HID;
    const __bf16* bptr1 = WhhT + (size_t)(nw + 16 + nlane) * HID;
    const __bf16* cptr0 = WhxT + (size_t)(nw + nlane) * IN_DIM;
    const __bf16* cptr1 = WhxT + (size_t)(nw + 16 + nlane) * IN_DIM;

    // cooperative staging: 8 threads/row, 32B each
    const int ldr = tid >> 3;
    const int ldc = (tid & 7) * 16;
    const __bf16* a_src = h_in + (size_t)(m0 + ldr) * HID + ldc;
    const __bf16* x_src = x_bf + (size_t)(m0 + ldr) * (SEQ * IN_DIM) + (size_t)t * IN_DIM + ldc;

    const unsigned lds_off0 = (unsigned)(uintptr_t)(&As[0][ldr * AROW + ldc]);
    const unsigned lds_off1 = (unsigned)(uintptr_t)(&As[1][ldr * AROW + ldc]);

    // prologue: stage chunk 0
    async_stage32(lds_off0, a_src);

    int cur = 0;
    for (int c = 0; c < NCHUNK_T; ++c) {
        wait_async0();       // own async loads for buf[cur] complete
        __syncthreads();     // all waves' staging complete; safe to reuse buf[cur^1]
        if (c + 1 < NCHUNK_T) {
            const int cn = c + 1;
            const __bf16* g = (cn < NCHUNK_H) ? (a_src + cn * CHUNK_K)
                                              : (x_src + (cn - NCHUNK_H) * CHUNK_K);
            async_stage32(cur ? lds_off0 : lds_off1, g);
        }

        const __bf16* Ab = As[cur];
        const __bf16* b0r;
        const __bf16* b1r;
        if (c < NCHUNK_H) { b0r = bptr0 + c * CHUNK_K;               b1r = bptr1 + c * CHUNK_K; }
        else              { b0r = cptr0 + (c - NCHUNK_H) * CHUNK_K;  b1r = cptr1 + (c - NCHUNK_H) * CHUNK_K; }

#pragma unroll
        for (int ks = 0; ks < CHUNK_K; ks += 32) {
            v16bf a0 = lds_a_frag(Ab, 0,  ks, lane);
            v16bf a1 = lds_a_frag(Ab, 16, ks, lane);
            v16bf b0 = glb_b_frag(b0r, ks, lane);
            v16bf b1 = glb_b_frag(b1r, ks, lane);
            acc[0][0] = __builtin_amdgcn_wmma_f32_16x16x32_bf16(false, a0, false, b0, (short)0, acc[0][0], false, false);
            acc[1][0] = __builtin_amdgcn_wmma_f32_16x16x32_bf16(false, a1, false, b0, (short)0, acc[1][0], false, false);
            acc[0][1] = __builtin_amdgcn_wmma_f32_16x16x32_bf16(false, a0, false, b1, (short)0, acc[0][1], false, false);
            acc[1][1] = __builtin_amdgcn_wmma_f32_16x16x32_bf16(false, a1, false, b1, (short)0, acc[1][1], false, false);
        }
        cur ^= 1;
    }

    // epilogue: C/D layout — lane l: n = l&15, m = r + 8*(l>>4)
    const int mo = (lane & 16) ? 8 : 0;
#pragma unroll
    for (int nt = 0; nt < 2; ++nt) {
        const int n = nw + nt * 16 + nlane;
        const float bh = Bh[n];
#pragma unroll
        for (int mt = 0; mt < 2; ++mt) {
#pragma unroll
            for (int r = 0; r < 8; ++r) {
                const int m = m0 + mt * 16 + mo + r;
                h_out[(size_t)m * HID + n] = (__bf16)tanhf(acc[mt][nt][r] + bh);
            }
        }
    }
}

// ---------------------------------------------------------------------------
// out = h_final @ Wph + Bp   [BATCH x HID] x [HID x CLS] -> f32 [BATCH x CLS]
// grid = (CLS/256, BATCH/32) = (4, 8)
// ---------------------------------------------------------------------------
__global__ void __launch_bounds__(256)
final_proj_kernel(const __bf16* __restrict__ h,     // [BATCH][HID]
                  const __bf16* __restrict__ WphT,  // [CLS][HID]
                  const float*  __restrict__ Bp,
                  float*        __restrict__ out) {
    __shared__ __bf16 As[2][32 * AROW];

    const int tid   = threadIdx.x;
    const int lane  = tid & 31;
    const int wave  = tid >> 5;
    const int nlane = lane & 15;
    const int m0    = blockIdx.y * 32;
    const int nw    = blockIdx.x * 256 + wave * 32;

    v8f acc[2][2] = {};

    const __bf16* bptr0 = WphT + (size_t)(nw + nlane) * HID;
    const __bf16* bptr1 = WphT + (size_t)(nw + 16 + nlane) * HID;

    const int ldr = tid >> 3;
    const int ldc = (tid & 7) * 16;
    const __bf16* a_src = h + (size_t)(m0 + ldr) * HID + ldc;

    const unsigned lds_off0 = (unsigned)(uintptr_t)(&As[0][ldr * AROW + ldc]);
    const unsigned lds_off1 = (unsigned)(uintptr_t)(&As[1][ldr * AROW + ldc]);

    async_stage32(lds_off0, a_src);

    int cur = 0;
    for (int c = 0; c < NCHUNK_H; ++c) {
        wait_async0();
        __syncthreads();
        if (c + 1 < NCHUNK_H)
            async_stage32(cur ? lds_off0 : lds_off1, a_src + (c + 1) * CHUNK_K);

        const __bf16* Ab = As[cur];
        const __bf16* b0r = bptr0 + c * CHUNK_K;
        const __bf16* b1r = bptr1 + c * CHUNK_K;

#pragma unroll
        for (int ks = 0; ks < CHUNK_K; ks += 32) {
            v16bf a0 = lds_a_frag(Ab, 0,  ks, lane);
            v16bf a1 = lds_a_frag(Ab, 16, ks, lane);
            v16bf b0 = glb_b_frag(b0r, ks, lane);
            v16bf b1 = glb_b_frag(b1r, ks, lane);
            acc[0][0] = __builtin_amdgcn_wmma_f32_16x16x32_bf16(false, a0, false, b0, (short)0, acc[0][0], false, false);
            acc[1][0] = __builtin_amdgcn_wmma_f32_16x16x32_bf16(false, a1, false, b0, (short)0, acc[1][0], false, false);
            acc[0][1] = __builtin_amdgcn_wmma_f32_16x16x32_bf16(false, a0, false, b1, (short)0, acc[0][1], false, false);
            acc[1][1] = __builtin_amdgcn_wmma_f32_16x16x32_bf16(false, a1, false, b1, (short)0, acc[1][1], false, false);
        }
        cur ^= 1;
    }

    const int mo = (lane & 16) ? 8 : 0;
#pragma unroll
    for (int nt = 0; nt < 2; ++nt) {
        const int n = nw + nt * 16 + nlane;
        const float bp = Bp[n];
#pragma unroll
        for (int mt = 0; mt < 2; ++mt) {
#pragma unroll
            for (int r = 0; r < 8; ++r) {
                const int m = m0 + mt * 16 + mo + r;
                out[(size_t)m * CLS + n] = acc[mt][nt][r] + bp;
            }
        }
    }
}

// ---------------------------------------------------------------------------
// Conversion / init kernels
// ---------------------------------------------------------------------------

// dst[n][k] = (bf16)src[k][n]   (src: [K][N] f32, dst: [N][K] bf16)
__global__ void cvt_transpose_kernel(const float* __restrict__ src, __bf16* __restrict__ dst,
                                     int K, int N) {
    int idx = blockIdx.x * blockDim.x + threadIdx.x;
    if (idx >= K * N) return;
    int n = idx / K;
    int k = idx - n * K;
    dst[idx] = (__bf16)src[(size_t)k * N + n];
}

// elementwise f32 -> bf16, 4 per thread
__global__ void cvt_bf16_kernel(const float* __restrict__ src, __bf16* __restrict__ dst, int n4) {
    int i = blockIdx.x * blockDim.x + threadIdx.x;
    if (i >= n4) return;
    float4 f = reinterpret_cast<const float4*>(src)[i];
    __bf16* d = dst + (size_t)i * 4;
    d[0] = (__bf16)f.x; d[1] = (__bf16)f.y; d[2] = (__bf16)f.z; d[3] = (__bf16)f.w;
}

__global__ void zero_bf16_kernel(__bf16* __restrict__ p, int n) {
    int i = blockIdx.x * blockDim.x + threadIdx.x;
    if (i < n) p[i] = (__bf16)0.0f;
}

// ---------------------------------------------------------------------------

extern "C" void kernel_launch(void* const* d_in, const int* in_sizes, int n_in,
                              void* d_out, int out_size, void* d_ws, size_t ws_size,
                              hipStream_t stream) {
    (void)in_sizes; (void)n_in; (void)out_size; (void)ws_size;
    const float* x   = (const float*)d_in[0];
    const float* Whx = (const float*)d_in[1];
    const float* Whh = (const float*)d_in[2];
    const float* Wph = (const float*)d_in[3];
    const float* Bh  = (const float*)d_in[4];
    const float* Bp  = (const float*)d_in[5];
    float* out = (float*)d_out;

    char* ws = (char*)d_ws;
    const size_t MB = 1u << 20;
    __bf16* WhxT = (__bf16*)(ws);             // 2048*256*2      = 1 MB
    __bf16* WhhT = (__bf16*)(ws + 1 * MB);    // 2048*2048*2     = 8 MB
    __bf16* WphT = (__bf16*)(ws + 9 * MB);    // 1024*2048*2     = 4 MB
    __bf16* h0   = (__bf16*)(ws + 13 * MB);   // 256*2048*2      = 1 MB
    __bf16* h1   = (__bf16*)(ws + 14 * MB);   // 256*2048*2      = 1 MB
    __bf16* x_bf = (__bf16*)(ws + 15 * MB);   // 256*512*256*2   = 64 MB  (ends at 79 MB)

    // One-time (per call) conversions
    cvt_transpose_kernel<<<(IN_DIM * HID + 255) / 256, 256, 0, stream>>>(Whx, WhxT, IN_DIM, HID);
    cvt_transpose_kernel<<<(HID * HID + 255) / 256, 256, 0, stream>>>(Whh, WhhT, HID, HID);
    cvt_transpose_kernel<<<(HID * CLS + 255) / 256, 256, 0, stream>>>(Wph, WphT, HID, CLS);
    {
        int n4 = (BATCH * SEQ * IN_DIM) / 4;
        cvt_bf16_kernel<<<(n4 + 255) / 256, 256, 0, stream>>>(x, x_bf, n4);
    }
    zero_bf16_kernel<<<(BATCH * HID + 255) / 256, 256, 0, stream>>>(h0, BATCH * HID);

    // Sequential recurrence: stream ordering is the per-step global barrier.
    dim3 sgrid(HID / 256, BATCH / 32);   // (8, 8)
    for (int t = 0; t < SEQ; ++t) {
        const __bf16* hin  = (t & 1) ? h1 : h0;
        __bf16*       hout = (t & 1) ? h0 : h1;
        rnn_step_kernel<<<sgrid, 256, 0, stream>>>(x_bf, WhxT, WhhT, Bh, hin, hout, t);
    }

    // SEQ even -> last step (t=511, odd) wrote h0.
    dim3 fgrid(CLS / 256, BATCH / 32);   // (4, 8)
    final_proj_kernel<<<fgrid, 256, 0, stream>>>(h0, WphT, Bp, out);
}
